// DecoderRNNWithAttention_80522046865911
// MI455X (gfx1250) — matrix operand order repaired and checked by reading, not compiled, not run on hardware
//
#include <hip/hip_runtime.h>

// ---------------------------------------------------------------------------
// DecoderRNNWithAttention — MI455X (gfx1250) implementation.
// Attention is degenerate (softmax over 1 pixel -> alpha==1, context==features),
// so the model is a plain LSTM:  gates = w_t@Wih_w.T + feat@Wih_e.T + b + h@Whh.T
// bf16 WMMA GEMMs with async global->LDS staging (gfx1250 ASYNCcnt path).
// The 25-step recurrence GEMM has a pure-store epilogue; the per-step gate
// base (Wpart[t]) is fused into the elementwise LSTM cell instead.
// ---------------------------------------------------------------------------

typedef __attribute__((ext_vector_type(16))) __bf16 v16bf;
typedef __attribute__((ext_vector_type(8)))  float  v8f;

#define Bsz   128
#define Tlen  25
#define Edim  512
#define Hdim  1024
#define Vdim  32000
#define ENCd  400
#define G4H   4096      // 4*H
#define XROW  912       // E + ENC (W_ih row length)

// ---- bf16 WMMA GEMM:  Out = A[M,K] @ W[N,K]^T (+Cadd[(r%CMOD),:]) (+bias[col])
// Output row r is remapped: t = r / ROWS_PER_T, b = r % ROWS_PER_T,
// Out[b*os_b + t*os_t + col]  (lets the fcn GEMM write [B,T,V] directly).
#define BM 64
#define BN 256
#define BK 32
#define LDSS 40   // padded row stride in bf16 elems (80B) -> conflict-free frags

union Frag { v16bf v; float4 f4[2]; };

// gfx1250 async global->LDS copy (tracked with ASYNCcnt), 16B per lane.
__device__ __forceinline__ void async_ld_b128(unsigned lds_off, const void* gaddr)
{
    asm volatile("global_load_async_to_lds_b128 %0, %1, off"
                 :: "v"(lds_off), "v"((unsigned long long)gaddr) : "memory");
}
__device__ __forceinline__ void wait_async0()
{
    asm volatile("s_wait_asynccnt 0x0" ::: "memory");
}

template <bool HAS_CADD, bool HAS_BIAS, int ROWS_PER_T, int CMOD>
__global__ __launch_bounds__(256) void gemm_bf16_wmma(
    const __bf16* __restrict__ A,     // [M,K] row-major bf16
    const __bf16* __restrict__ W,     // [N,K] row-major bf16
    const float*  __restrict__ Cadd,  // [CMOD,N] fp32 (if HAS_CADD)
    const float*  __restrict__ bias,  // [N] fp32 (if HAS_BIAS)
    float* __restrict__ Out,
    int M, int N, int K,
    long os_b, long os_t)
{
    __shared__ __align__(16) __bf16 sA[BM * LDSS];
    __shared__ __align__(16) __bf16 sB[BN * LDSS];

    const int tile_m = blockIdx.y * BM;
    const int tile_n = blockIdx.x * BN;
    const int tid  = threadIdx.x;
    const int lane = tid & 31;
    const int wave = tid >> 5;
    const int wm = (wave >> 2) * 32;   // 0 / 32             (2 wave-rows)
    const int wn = (wave &  3) * 64;   // 0 / 64 / 128 / 192 (4 wave-cols)
    const int l  = lane & 15;
    const int hi = lane >> 4;

    // staging assignment: r = tid/4 (row), s = tid%4 (16B segment of 64B row)
    const int sr = tid >> 2, ss = tid & 3;
    const unsigned aBase = (unsigned)(size_t)(&sA[0]);
    const unsigned bBase = (unsigned)(size_t)(&sB[0]);
    const unsigned aOff  = aBase + (unsigned)(sr * LDSS + ss * 8) * 2u;
    const unsigned bOff0 = bBase + (unsigned)(sr * LDSS + ss * 8) * 2u;

    v8f acc[2][4];
    const v8f vzero = {0.f, 0.f, 0.f, 0.f, 0.f, 0.f, 0.f, 0.f};
#pragma unroll
    for (int i = 0; i < 2; ++i)
#pragma unroll
        for (int j = 0; j < 4; ++j) acc[i][j] = vzero;

    for (int k0 = 0; k0 < K; k0 += BK) {
        // Stage A tile [64 x 32] + B tile [256 x 32] straight into LDS
        // via the gfx1250 async DMA path (no VGPR round-trip).
        async_ld_b128(aOff, A + (long)(tile_m + sr) * K + k0 + ss * 8);
#pragma unroll
        for (int h = 0; h < 4; ++h) {
            const int r = sr + h * 64;
            async_ld_b128(bOff0 + (unsigned)(h * 64 * LDSS) * 2u,
                          W + (long)(tile_n + r) * K + k0 + ss * 8);
        }
        wait_async0();
        __syncthreads();

        // A frag (16x32 bf16): lane<16 -> K{0-7,16-23}; lane>=16 -> K{8-15,24-31}
        Frag a[2], b[4];
#pragma unroll
        for (int i = 0; i < 2; ++i) {
            const __bf16* rp = &sA[(wm + i * 16 + l) * LDSS];
            a[i].f4[0] = *reinterpret_cast<const float4*>(rp + hi * 8);
            a[i].f4[1] = *reinterpret_cast<const float4*>(rp + 16 + hi * 8);
        }
        // B frag (32x16 bf16): lane<16 -> K{0-7,16-23} of col l; lane>=16 -> K{8-15,24-31}
#pragma unroll
        for (int j = 0; j < 4; ++j) {
            const __bf16* rp = &sB[(wn + j * 16 + l) * LDSS];
            b[j].f4[0] = *reinterpret_cast<const float4*>(rp + hi * 8);
            b[j].f4[1] = *reinterpret_cast<const float4*>(rp + 16 + hi * 8);
        }
#pragma unroll
        for (int i = 0; i < 2; ++i)
#pragma unroll
            for (int j = 0; j < 4; ++j)
                acc[i][j] = __builtin_amdgcn_wmma_f32_16x16x32_bf16(
                    false, a[i].v, false, b[j].v, (short)0, acc[i][j], false, false);
        __syncthreads();
    }

    // Epilogue. C/D layout: lanes 0-15 rows m..m+7, lanes 16-31 rows m+8..m+15.
    // row0 is a multiple of 8 and ROWS_PER_T/CMOD are multiples of 8, so the
    // 8-row span never wraps: remap once per fragment, then walk pointers.
#pragma unroll
    for (int i = 0; i < 2; ++i) {
#pragma unroll
        for (int j = 0; j < 4; ++j) {
            const int colg = tile_n + wn + j * 16 + l;
            const int row0 = tile_m + wm + i * 16 + hi * 8;
            const float bv = HAS_BIAS ? bias[colg] : 0.f;
            float add[8];
            if (HAS_CADD) {
                const float* cp = Cadd + (long)(row0 % CMOD) * N + colg;
#pragma unroll
                for (int v = 0; v < 8; ++v) { add[v] = bv + *cp; cp += N; }
            } else {
#pragma unroll
                for (int v = 0; v < 8; ++v) add[v] = bv;
            }
            const int tt  = row0 / ROWS_PER_T;
            const int bb0 = row0 % ROWS_PER_T;
            float* op = Out + (long)bb0 * os_b + (long)tt * os_t + colg;
#pragma unroll
            for (int v = 0; v < 8; ++v) { *op = acc[i][j][v] + add[v]; op += os_b; }
        }
    }
}

// ---- fp32 -> bf16 strided row extract (used for W_ih[:, :E], W_hh, fcn_W)
__global__ void cvt_bf16_kernel(const float* __restrict__ src, __bf16* __restrict__ dst,
                                long total, int cols, long src_stride, long src_off)
{
    long idx = (long)blockIdx.x * blockDim.x + threadIdx.x;
    if (idx >= total) return;
    long r = idx / cols, c = idx % cols;
    dst[idx] = (__bf16)src[r * src_stride + src_off + c];
}

// ---- gather teacher-forced word embeddings -> bf16 [T*B, E]
__global__ void words_kernel(const float* __restrict__ emb_W,
                             const int* __restrict__ captions,
                             __bf16* __restrict__ wordsBF)
{
    const int row = blockIdx.x;              // 0 .. T*B-1
    const int t = row / Bsz, b = row % Bsz;
    const int tok = (t == 0) ? 1 : captions[b * Tlen + (t - 1)];
    const float* src = emb_W + (long)tok * Edim;
    for (int c = threadIdx.x; c < Edim; c += blockDim.x)
        wordsBF[(long)row * Edim + c] = (__bf16)src[c];
}

// ---- Cbase[b,j] = features[b,:]@Wih_enc[j,:].T + b_ih[j] + b_hh[j]  (runs once)
__global__ void cbase_kernel(const float* __restrict__ features,
                             const float* __restrict__ W_ih,
                             const float* __restrict__ b_ih,
                             const float* __restrict__ b_hh,
                             float* __restrict__ Cbase)
{
    const int j = blockIdx.x * blockDim.x + threadIdx.x;   // 0..4095
    const int b = blockIdx.y;                              // 0..127
    if (j >= G4H) return;
    const float* wrow = W_ih + (long)j * XROW + Edim;
    const float* f = features + (long)b * ENCd;
    float s = b_ih[j] + b_hh[j];
#pragma unroll 4
    for (int k = 0; k < ENCd; ++k) s += f[k] * wrow[k];
    Cbase[(long)b * G4H + j] = s;
}

// ---- zero h (bf16) and c (fp32)
__global__ void init_state_kernel(__bf16* __restrict__ h_bf, float* __restrict__ c)
{
    const int i = blockIdx.x * blockDim.x + threadIdx.x;
    if (i < Bsz * Hdim) { h_bf[i] = (__bf16)0.f; c[i] = 0.f; }
}

__device__ __forceinline__ float sigmoidf_(float x) { return 1.f / (1.f + __expf(-x)); }

// ---- elementwise LSTM cell: gates_full = gates + Wpart[t]; update c, h.
// All streams (gates, Wpart row, c) are fully coalesced.
__global__ void lstm_cell_kernel(const float* __restrict__ gates,   // [128,4096] h@Whh.T
                                 const float* __restrict__ wpart_t, // [128,4096] word+feat+bias
                                 float* __restrict__ c,
                                 __bf16* __restrict__ h_bf,
                                 __bf16* __restrict__ H_all,
                                 int t)
{
    const int i = blockIdx.x * blockDim.x + threadIdx.x;   // 0 .. B*H-1
    if (i >= Bsz * Hdim) return;
    const int b = i >> 10, k = i & (Hdim - 1);
    const long base = (long)b * G4H;
    const float ig = gates[base + k]            + wpart_t[base + k];
    const float fg = gates[base + k + Hdim]     + wpart_t[base + k + Hdim];
    const float gg = gates[base + k + 2 * Hdim] + wpart_t[base + k + 2 * Hdim];
    const float og = gates[base + k + 3 * Hdim] + wpart_t[base + k + 3 * Hdim];
    const float cn = sigmoidf_(fg) * c[i] + sigmoidf_(ig) * tanhf(gg);
    const float hn = sigmoidf_(og) * tanhf(cn);
    c[i] = cn;
    const __bf16 hb = (__bf16)hn;
    h_bf[i] = hb;
    if (t >= 1) H_all[(long)(t - 1) * (Bsz * Hdim) + i] = hb;
}

extern "C" void kernel_launch(void* const* d_in, const int* in_sizes, int n_in,
                              void* d_out, int out_size, void* d_ws, size_t ws_size,
                              hipStream_t stream)
{
    const float* features = (const float*)d_in[0];   // [128,400]
    const int*   captions = (const int*)  d_in[1];   // [128,25]
    const float* emb_W    = (const float*)d_in[2];   // [32000,512]
    const float* W_ih     = (const float*)d_in[3];   // [4096,912]
    const float* W_hh     = (const float*)d_in[4];   // [4096,1024]
    const float* b_ih     = (const float*)d_in[5];   // [4096]
    const float* b_hh     = (const float*)d_in[6];   // [4096]
    const float* fcn_W    = (const float*)d_in[13];  // [32000,1024]
    const float* fcn_b    = (const float*)d_in[14];  // [32000]
    float* out = (float*)d_out;                      // [128,24,32000]

    // ---- carve workspace (256B aligned)
    size_t off = 0;
    auto carve = [&](size_t bytes) -> void* {
        off = (off + 255) & ~(size_t)255;
        void* p = (char*)d_ws + off;
        off += bytes;
        return p;
    };
    __bf16* wordsBF  = (__bf16*)carve((size_t)Tlen * Bsz * Edim * 2); // [3200,512]
    __bf16* wihw_bf  = (__bf16*)carve((size_t)G4H * Edim * 2);        // [4096,512]
    __bf16* whh_bf   = (__bf16*)carve((size_t)G4H * Hdim * 2);        // [4096,1024]
    __bf16* fcn_bf   = (__bf16*)carve((size_t)Vdim * Hdim * 2);       // [32000,1024]
    float*  Cbase    = (float*) carve((size_t)Bsz * G4H * 4);         // [128,4096]
    float*  Wpart    = (float*) carve((size_t)Tlen * Bsz * G4H * 4);  // [3200,4096]
    float*  gates    = (float*) carve((size_t)Bsz * G4H * 4);         // [128,4096]
    __bf16* h_bf     = (__bf16*)carve((size_t)Bsz * Hdim * 2);        // [128,1024]
    float*  c_st     = (float*) carve((size_t)Bsz * Hdim * 4);        // [128,1024]
    __bf16* H_all    = (__bf16*)carve((size_t)(Tlen - 1) * Bsz * Hdim * 2); // [3072,1024]
    (void)ws_size; (void)in_sizes; (void)n_in; (void)out_size;

    // ---- one-time precompute
    {   // W_ih word half -> bf16 (rows 4096, take cols [0,512) of stride-912 rows)
        long tot = (long)G4H * Edim;
        cvt_bf16_kernel<<<(tot + 255) / 256, 256, 0, stream>>>(W_ih, wihw_bf, tot, Edim, XROW, 0);
    }
    {   // W_hh -> bf16
        long tot = (long)G4H * Hdim;
        cvt_bf16_kernel<<<(tot + 255) / 256, 256, 0, stream>>>(W_hh, whh_bf, tot, Hdim, Hdim, 0);
    }
    {   // fcn_W -> bf16
        long tot = (long)Vdim * Hdim;
        cvt_bf16_kernel<<<(tot + 255) / 256, 256, 0, stream>>>(fcn_W, fcn_bf, tot, Hdim, Hdim, 0);
    }
    words_kernel<<<Tlen * Bsz, 256, 0, stream>>>(emb_W, captions, wordsBF);
    cbase_kernel<<<dim3(G4H / 256, Bsz), 256, 0, stream>>>(features, W_ih, b_ih, b_hh, Cbase);

    // Wpart[t*128+b, :] = words @ Wih_word.T + Cbase[b,:]   (M=3200,N=4096,K=512)
    gemm_bf16_wmma<true, false, Tlen * Bsz, Bsz>
        <<<dim3(G4H / BN, (Tlen * Bsz) / BM), 256, 0, stream>>>(
        wordsBF, wihw_bf, Cbase, nullptr, Wpart,
        Tlen * Bsz, G4H, Edim, /*os_b=*/G4H, /*os_t=*/0);

    init_state_kernel<<<(Bsz * Hdim + 255) / 256, 256, 0, stream>>>(h_bf, c_st);

    // ---- sequential recurrence (25 steps); gates GEMM epilogue is pure stores
    for (int t = 0; t < Tlen; ++t) {
        // gates = h @ Whh.T   (M=128,N=4096,K=1024)
        gemm_bf16_wmma<false, false, Bsz, 1>
            <<<dim3(G4H / BN, Bsz / BM), 256, 0, stream>>>(
            h_bf, whh_bf, nullptr, nullptr, gates,
            Bsz, G4H, Hdim, /*os_b=*/G4H, /*os_t=*/0);
        lstm_cell_kernel<<<(Bsz * Hdim + 255) / 256, 256, 0, stream>>>(
            gates, Wpart + (long)t * Bsz * G4H, c_st, h_bf, H_all, t);
    }

    // ---- batched logits: [3072,1024] @ [1024,32000]^T + fcn_b -> d_out [B,24,V]
    gemm_bf16_wmma<false, true, Bsz, 1>
        <<<dim3(Vdim / BN, ((Tlen - 1) * Bsz) / BM), 256, 0, stream>>>(
        H_all, fcn_bf, nullptr, fcn_b, out,
        (Tlen - 1) * Bsz, Vdim, Hdim,
        /*os_b=*/(long)(Tlen - 1) * Vdim, /*os_t=*/Vdim);
}
